// GaussianRefinementBlock_14568529068108
// MI455X (gfx1250) — matrix operand order repaired and checked by reading, not compiled, not run on hardware
//
#include <hip/hip_runtime.h>
#include <cstdint>

typedef __attribute__((ext_vector_type(16))) _Float16 v16h;
typedef __attribute__((ext_vector_type(8)))  _Float16 v8h;
typedef __attribute__((ext_vector_type(8)))  float    v8f;

#define DINL __device__ __forceinline__

// ---------------- problem constants (match reference setup) ----------------
#define NQ     1024
#define CCH    128
#define NCAM   6
#define STOT   14960      // 64*176 + 32*88 + 16*44 + 8*22
#define GRID3  32768      // 32^3
#define PROPD  28         // 3+3+4+1+17
#define KCONV  3456       // 27*128
#define MSROWS 89760      // NCAM*STOT
#define IMGW   704.0f
#define IMGH   256.0f

#ifndef __has_builtin
#define __has_builtin(x) 0
#endif
#if __has_builtin(__builtin_amdgcn_global_load_async_to_lds_b128)
#define HAVE_ASYNC_B128 1
#else
#define HAVE_ASYNC_B128 0
#endif
#if __has_builtin(__builtin_amdgcn_s_wait_asynccnt)
#define HAVE_WAIT_ASYNC 1
#else
#define HAVE_WAIT_ASYNC 0
#endif

// exact pointee type the builtin expects: GCC-style 4x i32 vector
typedef int v4i_ __attribute__((vector_size(16)));
typedef __attribute__((address_space(1))) v4i_ gv4i;   // prints as "__device__" in HIP
typedef __attribute__((address_space(3))) v4i_ lv4i;   // prints as "__shared__" in HIP

// Async copy 16B from global to LDS (per active lane). CDNA5 ASYNCcnt path.
DINL void stage_async_b128(const _Float16* gsrc, _Float16* ldst) {
#if HAVE_ASYNC_B128
  __builtin_amdgcn_global_load_async_to_lds_b128(
      (gv4i*)(uintptr_t)gsrc,
      (lv4i*)(unsigned int)(uintptr_t)ldst, 0, 0);
#else
  unsigned int loff = (unsigned int)(uintptr_t)ldst;
  unsigned long long ga = (unsigned long long)(uintptr_t)gsrc;
  asm volatile("global_load_async_to_lds_b128 %0, %1, off"
               :: "v"(loff), "v"(ga) : "memory");
#endif
}

DINL void wait_async_le1() {
#if HAVE_WAIT_ASYNC
  __builtin_amdgcn_s_wait_asynccnt(1);
#else
  asm volatile("s_wait_asynccnt 0x1" ::: "memory");
#endif
}
DINL void wait_async_le0() {
#if HAVE_WAIT_ASYNC
  __builtin_amdgcn_s_wait_asynccnt(0);
#else
  asm volatile("s_wait_asynccnt 0x0" ::: "memory");
#endif
}

DINL float gelu_f(float x) {
  return 0.5f * x * (1.0f + tanhf(0.7978845608028654f * (x + 0.044715f * x * x * x)));
}

// ---------------- utility kernels ----------------
__global__ void k_fill_zero(float* p, long long n) {
  long long i = (long long)blockIdx.x * blockDim.x + threadIdx.x;
  if (i < n) p[i] = 0.f;
}

__global__ void k_f32_to_f16(const float* __restrict__ s, _Float16* __restrict__ d, long long n) {
  long long i = (long long)blockIdx.x * blockDim.x + threadIdx.x;
  if (i < n) d[i] = (_Float16)s[i];
}

// dst[n*K+k] = src[k*N+n]  (weights f32 KxN row-major -> f16 NxK "Bt")
__global__ void k_transpose_f16(const float* __restrict__ s, _Float16* __restrict__ d, int K, int N) {
  int i = blockIdx.x * blockDim.x + threadIdx.x;
  if (i >= K * N) return;
  int k = i / N, n = i - k * N;
  d[(long long)n * K + k] = (_Float16)s[i];
}

// ---------------- stage 1: voxel scatter ----------------
DINL void voxel_of(const float* props, int n, int& vx, int& vy, int& vz) {
  vx = min(max((int)floorf((props[n * PROPD + 0] + 4.f) * 4.f), 0), 31);
  vy = min(max((int)floorf((props[n * PROPD + 1] + 4.f) * 4.f), 0), 31);
  vz = min(max((int)floorf((props[n * PROPD + 2] + 4.f) * 4.f), 0), 31);
}

__global__ void k_scatter(const float* __restrict__ q, const float* __restrict__ props,
                          float* __restrict__ grid, float* __restrict__ cnt) {
  int n = blockIdx.x, c = threadIdx.x;
  int vx, vy, vz; voxel_of(props, n, vx, vy, vz);
  int flat = (vx * 32 + vy) * 32 + vz;
  atomicAdd(&grid[(long long)flat * CCH + c], q[(long long)n * CCH + c]);
  if (c == 0) atomicAdd(&cnt[flat], 1.f);
}

// ---------------- stage 2: gather 27-neighborhood -> sparse-conv GEMM A (f16) ----------------
__global__ void k_gather_convA(const float* __restrict__ props, const float* __restrict__ grid,
                               const float* __restrict__ cnt, _Float16* __restrict__ A) {
  int n = blockIdx.x, c = threadIdx.x;
  int vx, vy, vz; voxel_of(props, n, vx, vy, vz);
  for (int nb = 0; nb < 27; ++nb) {
    int kd = nb / 9, kh = (nb / 3) % 3, kw = nb % 3;
    int x = vx + kd - 1, y = vy + kh - 1, z = vz + kw - 1;
    float v = 0.f;
    if ((unsigned)x < 32u && (unsigned)y < 32u && (unsigned)z < 32u) {
      int f = (x * 32 + y) * 32 + z;
      v = grid[(long long)f * CCH + c] / fmaxf(cnt[f], 1.f);
    }
    A[(long long)n * KCONV + nb * CCH + c] = (_Float16)v;
  }
}

// ---------------- WMMA GEMM: C(MxN,f32) = A(MxK,f16) * Bt(NxK,f16)^T + bias ----------------
// block = 128 threads (4 waves); wave w -> 16x16 D tile at (m0, n0+16w). grid=(M/16, N/64).
// A tile (16x32 f16, 1KB) staged via GLOBAL_LOAD_ASYNC_TO_LDS_B128 (threads 0..63, 16B each),
// double-buffered, synchronized with s_wait_asynccnt + barrier; waves read frags via ds_load_b128.
// B fragments loaded directly (per-wave unique, L2-resident weights).
// Fragment layouts per CDNA5 ISA 7.12.2:
//   A lane: m = lane&15; halves[0..7]=A[m][k+8*hi .. +7], halves[8..15]=A[m][k+8*hi+16 .. +23]
//   B lane: n = lane&15; halves[0..7]=B[k+8*hi .. +7][n], halves[8..15]=B[k+8*hi+16 .. +23][n]  (Bt = NxK)
//   D lane: acc[r] = D[r + 8*hi][lane&15]
__global__ void k_wmma_gemm(const _Float16* __restrict__ A, const _Float16* __restrict__ Bt,
                            const float* __restrict__ bias, float* __restrict__ C,
                            _Float16* __restrict__ C16, int M, int N, int K, int act) {
  __shared__ __attribute__((aligned(16))) _Float16 Abuf[2][16 * 32];
  int t = threadIdx.x;
  int wave = t >> 5, lane = t & 31;
  int m0 = blockIdx.x << 4;
  int n0 = (blockIdx.y << 6) + (wave << 4);
  int lo = lane & 15;
  int hi = lane >> 4;

  // staging role: thread t<64 copies 8 halves (16B): row sm, k-offset sk
  int sm = t >> 2;
  int sk = (t & 3) << 3;
  const _Float16* Asrc = A + (long long)(m0 + sm) * K + sk;

  const _Float16* Bp = Bt + (long long)(n0 + lo) * K + hi * 8;
  int afrag = lo * 32 + hi * 8;   // LDS offset of A fragment low chunk

  if (t < 64) stage_async_b128(Asrc, &Abuf[0][sm * 32 + sk]);

  v8f acc = {};
  int pb = 0;
  for (int k = 0; k < K; k += 32, pb ^= 1) {
    if (k + 32 < K) {
      if (t < 64) stage_async_b128(Asrc + k + 32, &Abuf[pb ^ 1][sm * 32 + sk]);
      __builtin_prefetch(Bp + k + 32, 0, 1);
      wait_async_le1();
    } else {
      wait_async_le0();
    }
    __syncthreads();
    v8h b0 = *(const v8h*)(Bp + k);
    v8h b1 = *(const v8h*)(Bp + k + 16);
    v8h a0 = *(const v8h*)&Abuf[pb][afrag];        // ds_load_b128
    v8h a1 = *(const v8h*)&Abuf[pb][afrag + 16];   // ds_load_b128
    v16h av, bv;
#pragma unroll
    for (int i = 0; i < 8; ++i) { av[i] = a0[i]; av[i + 8] = a1[i]; bv[i] = b0[i]; bv[i + 8] = b1[i]; }
    acc = __builtin_amdgcn_wmma_f32_16x16x32_f16(false, av, false, bv, (short)0, acc, false, false);
    __syncthreads();
  }
  int n = n0 + lo;
  float bb = bias ? bias[n] : 0.f;
#pragma unroll
  for (int r = 0; r < 8; ++r) {
    int m = m0 + r + (hi << 3);
    float v = acc[r] + bb;
    if (act) v = gelu_f(v);
    C[(long long)m * N + n] = v;
    if (C16) C16[(long long)m * N + n] = (_Float16)v;
  }
}

// ---------------- LayerNorm(base + delta) over C=128, optional f16 dual-store ----------------
__global__ void k_layernorm_res(const float* __restrict__ base, const float* __restrict__ delta,
                                const float* __restrict__ g, const float* __restrict__ b,
                                float* __restrict__ out, _Float16* __restrict__ out16) {
  __shared__ float sm[CCH];
  int row = blockIdx.x, c = threadIdx.x;
  long long idx = (long long)row * CCH + c;
  float x = base[idx] + (delta ? delta[idx] : 0.f);
  sm[c] = x; __syncthreads();
  for (int s = 64; s > 0; s >>= 1) { if (c < s) sm[c] += sm[c + s]; __syncthreads(); }
  float mean = sm[0] * (1.f / CCH); __syncthreads();
  float d = x - mean;
  sm[c] = d * d; __syncthreads();
  for (int s = 64; s > 0; s >>= 1) { if (c < s) sm[c] += sm[c + s]; __syncthreads(); }
  float var = sm[0] * (1.f / CCH);
  float y = d * rsqrtf(var + 1e-5f) * g[c] + b[c];
  out[idx] = y;
  if (out16) out16[idx] = (_Float16)y;
}

// ---------------- softmax over rows of length 128 (in-place) ----------------
__global__ void k_softmax128(float* __restrict__ x) {
  __shared__ float sm[CCH];
  int row = blockIdx.x, c = threadIdx.x;
  long long idx = (long long)row * CCH + c;
  float v = x[idx];
  sm[c] = v; __syncthreads();
  for (int s = 64; s > 0; s >>= 1) { if (c < s) sm[c] = fmaxf(sm[c], sm[c + s]); __syncthreads(); }
  float mx = sm[0]; __syncthreads();
  float e = __expf(v - mx);
  sm[c] = e; __syncthreads();
  for (int s = 64; s > 0; s >>= 1) { if (c < s) sm[c] += sm[c + s]; __syncthreads(); }
  x[idx] = e / sm[0];
}

// ---------------- ref_off: small 128->12 projection ----------------
__global__ void k_refoff(const float* __restrict__ q1, const float* __restrict__ w,
                         const float* __restrict__ b, float* __restrict__ out) {
  int n = blockIdx.x, j = threadIdx.x;
  if (j >= 12) return;
  float s = b[j];
  for (int c = 0; c < CCH; ++c) s += q1[(long long)n * CCH + c] * w[c * 12 + j];
  out[n * 12 + j] = s;
}

// ---------------- geometry: project, depth-weight mask, 2D reference ----------------
__global__ void k_geometry(const float* __restrict__ props, const float* __restrict__ refoff,
                           const float* __restrict__ l2i, const float* __restrict__ depth,
                           float* __restrict__ wmask, float* __restrict__ r2u, float* __restrict__ r2v) {
  int idx = blockIdx.x * blockDim.x + threadIdx.x;
  if (idx >= NCAM * NQ * 4) return;
  int r = idx & 3;
  int n = (idx >> 2) & (NQ - 1);
  int cam = idx >> 12;
  float px = props[n * PROPD + 0] + refoff[n * 12 + r * 3 + 0];
  float py = props[n * PROPD + 1] + refoff[n * 12 + r * 3 + 1];
  float pz = props[n * PROPD + 2] + refoff[n * 12 + r * 3 + 2];
  const float* M = l2i + cam * 16;
  float p0 = M[0] * px + M[1] * py + M[2]  * pz + M[3];
  float p1 = M[4] * px + M[5] * py + M[6]  * pz + M[7];
  float p2 = M[8] * px + M[9] * py + M[10] * pz + M[11];
  float z = p2;
  float zc = fmaxf(z, 1e-5f);
  float u = p0 / zc, v = p1 / zc;
  bool valid = (z > 1e-5f) && (u >= 0.f) && (u < IMGW) && (v >= 0.f) && (v < IMGH);
  int iu = min(max((int)(u * (176.f / IMGW)), 0), 175);
  int iv = min(max((int)(v * (64.f / IMGH)), 0), 63);
  float ds = depth[cam * (64 * 176) + iv * 176 + iu];
  float wm = __expf(-fabsf(z - ds)) * (valid ? 1.f : 0.f);
  int o = (cam * NQ + n) * 4 + r;
  wmask[o] = wm;
  r2u[o] = u / IMGW;
  r2v[o] = v / IMGH;
}

// ---------------- deformable attention bilinear gather ----------------
__global__ void k_deform_sample(const float* __restrict__ value, const float* __restrict__ attn,
                                const float* __restrict__ offs, const float* __restrict__ wmask,
                                const float* __restrict__ r2u, const float* __restrict__ r2v,
                                float* __restrict__ out) {
  const int Ls[4] = {0, 11264, 14080, 14784};
  const int Hs[4] = {64, 32, 16, 8};
  const int Wz[4] = {176, 88, 44, 22};
  int n = blockIdx.x;
  int t = threadIdx.x;
  int h = t >> 4, ch = t & 15;
  float acc = 0.f;
  for (int cam = 0; cam < NCAM; ++cam) {
    for (int r = 0; r < 4; ++r) {
      int go = (cam * NQ + n) * 4 + r;
      float wm = wmask[go];
      if (wm == 0.f) continue;  // uniform across block
      float ru = r2u[go], rv = r2v[go];
      for (int l = 0; l < 4; ++l) {
        int Wl = Wz[l], Hl = Hs[l];
        const float* vb = value + ((long long)cam * STOT + Ls[l]) * CCH + h * 16 + ch;
        float invW = 1.f / (float)Wl, invH = 1.f / (float)Hl;
        for (int p = 0; p < 8; ++p) {
          long long ob = (((((long long)n * 8 + h) * 4 + r) * 4 + l) * 8 + p) * 2;
          float locx = ru + offs[ob + 0] * invW;
          float locy = rv + offs[ob + 1] * invH;
          float x = locx * Wl - 0.5f;
          float y = locy * Hl - 0.5f;
          float x0f = floorf(x), y0f = floorf(y);
          float wx = x - x0f, wy = y - y0f;
          int x0 = min(max((int)x0f, 0), Wl - 1);
          int x1 = min(max((int)x0f + 1, 0), Wl - 1);
          int y0 = min(max((int)y0f, 0), Hl - 1);
          int y1 = min(max((int)y0f + 1, 0), Hl - 1);
          float v00 = vb[(long long)(y0 * Wl + x0) * CCH];
          float v10 = vb[(long long)(y0 * Wl + x1) * CCH];
          float v01 = vb[(long long)(y1 * Wl + x0) * CCH];
          float v11 = vb[(long long)(y1 * Wl + x1) * CCH];
          float samp = (1.f - wx) * (1.f - wy) * v00 + wx * (1.f - wy) * v10 +
                       (1.f - wx) * wy * v01 + wx * wy * v11;
          float aw = attn[((long long)n * 8 + h) * 128 + (r * 4 + l) * 8 + p];
          acc += aw * wm * samp;
        }
      }
    }
  }
  out[(long long)n * CCH + t] = acc;
}

// ---------------- props head: 128 -> 28, residual on gaussian_props ----------------
__global__ void k_props_head(const float* __restrict__ tbuf, const float* __restrict__ w,
                             const float* __restrict__ b, const float* __restrict__ gp,
                             float* __restrict__ out) {
  int n = blockIdx.x, j = threadIdx.x;
  if (j >= PROPD) return;
  float s = b[j];
  for (int c = 0; c < CCH; ++c) s += tbuf[(long long)n * CCH + c] * w[c * PROPD + j];
  out[(long long)n * PROPD + j] = gp[(long long)n * PROPD + j] + s;
}

// ---------------- workspace layout ----------------
static constexpr size_t alup(size_t x) { return (x + 255) & ~(size_t)255; }

// region A (recycled after conv GEMM)
static constexpr size_t O_GRID   = 0;
static constexpr size_t O_CNT    = alup(O_GRID + (size_t)GRID3 * CCH * 4);
static constexpr size_t O_ACONV  = alup(O_CNT + (size_t)GRID3 * 4);
static constexpr size_t O_REGA_E = alup(O_ACONV + (size_t)NQ * KCONV * 2);
// region A overlays (used only after conv GEMM completes)
static constexpr size_t O_OUTATT = 0;
static constexpr size_t O_OUTH16 = alup(O_OUTATT + (size_t)NQ * CCH * 4);
static constexpr size_t O_ATTOUT = alup(O_OUTH16 + (size_t)NQ * CCH * 2);
static constexpr size_t O_Q2     = alup(O_ATTOUT + (size_t)NQ * CCH * 4);
static constexpr size_t O_Q2H    = alup(O_Q2 + (size_t)NQ * CCH * 4);
static constexpr size_t O_HBUF   = alup(O_Q2H + (size_t)NQ * CCH * 2);
static constexpr size_t O_HBUF16 = alup(O_HBUF + (size_t)NQ * 512 * 4);
static constexpr size_t O_FFN    = alup(O_HBUF16 + (size_t)NQ * 512 * 2);
static constexpr size_t O_Q3H    = alup(O_FFN + (size_t)NQ * CCH * 4);
static constexpr size_t O_TBUF   = alup(O_Q3H + (size_t)NQ * CCH * 2);
static constexpr size_t O_TBUF16 = alup(O_TBUF + (size_t)NQ * CCH * 4);
static constexpr size_t O_OVL_E  = alup(O_TBUF16 + (size_t)NQ * CCH * 2);
static_assert(O_OVL_E <= O_REGA_E, "overlay must fit region A");
// f16 transposed weights
static constexpr size_t O_WSCT   = O_REGA_E;
static constexpr size_t O_OFFWT  = alup(O_WSCT + (size_t)KCONV * CCH * 2);
static constexpr size_t O_AWWT   = alup(O_OFFWT + (size_t)CCH * 2048 * 2);
static constexpr size_t O_VALWT  = alup(O_AWWT + (size_t)CCH * 1024 * 2);
static constexpr size_t O_OUTWT  = alup(O_VALWT + (size_t)CCH * CCH * 2);
static constexpr size_t O_F1T    = alup(O_OUTWT + (size_t)CCH * CCH * 2);
static constexpr size_t O_F2T    = alup(O_F1T + (size_t)CCH * 512 * 2);
static constexpr size_t O_P1T    = alup(O_F2T + (size_t)512 * CCH * 2);
// ms_feats f16 (recycled for offs/attn after value GEMM)
static constexpr size_t O_MSF16  = alup(O_P1T + (size_t)CCH * CCH * 2);
static constexpr size_t O_MSF_E  = alup(O_MSF16 + (size_t)MSROWS * CCH * 2);
static constexpr size_t O_OFFS   = O_MSF16;
static constexpr size_t O_ATTN   = alup(O_OFFS + (size_t)NQ * 2048 * 4);
static_assert(alup(O_ATTN + (size_t)NQ * 1024 * 4) <= O_MSF_E, "offs/attn overlay fits");
// persistent tail
static constexpr size_t O_VALUE  = O_MSF_E;
static constexpr size_t O_CONVO  = alup(O_VALUE + (size_t)MSROWS * CCH * 4);
static constexpr size_t O_Q1     = alup(O_CONVO + (size_t)NQ * CCH * 4);
static constexpr size_t O_Q1H    = alup(O_Q1 + (size_t)NQ * CCH * 4);
static constexpr size_t O_REFOFF = alup(O_Q1H + (size_t)NQ * CCH * 2);
static constexpr size_t O_WMASK  = alup(O_REFOFF + (size_t)NQ * 12 * 4);
static constexpr size_t O_R2U    = alup(O_WMASK + (size_t)NCAM * NQ * 4 * 4);
static constexpr size_t O_R2V    = alup(O_R2U + (size_t)NCAM * NQ * 4 * 4);
static constexpr size_t WS_NEED  = alup(O_R2V + (size_t)NCAM * NQ * 4 * 4);

extern "C" void kernel_launch(void* const* d_in, const int* in_sizes, int n_in,
                              void* d_out, int out_size, void* d_ws, size_t ws_size,
                              hipStream_t stream) {
  (void)in_sizes; (void)n_in; (void)out_size; (void)ws_size;
  const float* q      = (const float*)d_in[0];
  const float* gprops = (const float*)d_in[1];
  const float* msf    = (const float*)d_in[2];
  const float* depth  = (const float*)d_in[3];
  const float* l2i    = (const float*)d_in[4];
  // d_in[5]=img_h, d_in[6]=img_w (compile-time constants 256/704 per reference)
  const float* Wsc  = (const float*)d_in[7];
  const float* bsc  = (const float*)d_in[8];
  const float* scg  = (const float*)d_in[9];
  const float* scb  = (const float*)d_in[10];
  const float* row  = (const float*)d_in[11];  // ref_off_w
  const float* rob  = (const float*)d_in[12];  // ref_off_b
  const float* offw = (const float*)d_in[13];
  const float* offb = (const float*)d_in[14];
  const float* aww  = (const float*)d_in[15];
  const float* awb  = (const float*)d_in[16];
  const float* valw = (const float*)d_in[17];
  const float* valb = (const float*)d_in[18];
  const float* outw = (const float*)d_in[19];
  const float* outb = (const float*)d_in[20];
  const float* ang  = (const float*)d_in[21];
  const float* anb  = (const float*)d_in[22];
  const float* f1w  = (const float*)d_in[23];
  const float* f1b  = (const float*)d_in[24];
  const float* f2w  = (const float*)d_in[25];
  const float* f2b  = (const float*)d_in[26];
  const float* fng  = (const float*)d_in[27];
  const float* fnb  = (const float*)d_in[28];
  const float* p1w  = (const float*)d_in[29];
  const float* p1b  = (const float*)d_in[30];
  const float* p2w  = (const float*)d_in[31];
  const float* p2b  = (const float*)d_in[32];

  char* ws = (char*)d_ws;
  float*    grid_f  = (float*)(ws + O_GRID);
  float*    cnt_f   = (float*)(ws + O_CNT);
  _Float16* Aconv   = (_Float16*)(ws + O_ACONV);
  _Float16* wsct    = (_Float16*)(ws + O_WSCT);
  _Float16* offwt   = (_Float16*)(ws + O_OFFWT);
  _Float16* awwt    = (_Float16*)(ws + O_AWWT);
  _Float16* valwt   = (_Float16*)(ws + O_VALWT);
  _Float16* outwt   = (_Float16*)(ws + O_OUTWT);
  _Float16* f1t     = (_Float16*)(ws + O_F1T);
  _Float16* f2t     = (_Float16*)(ws + O_F2T);
  _Float16* p1t     = (_Float16*)(ws + O_P1T);
  _Float16* msf16   = (_Float16*)(ws + O_MSF16);
  float*    value   = (float*)(ws + O_VALUE);
  float*    convout = (float*)(ws + O_CONVO);
  float*    q1      = (float*)(ws + O_Q1);
  _Float16* q1h     = (_Float16*)(ws + O_Q1H);
  float*    refoff  = (float*)(ws + O_REFOFF);
  float*    wmask   = (float*)(ws + O_WMASK);
  float*    r2u     = (float*)(ws + O_R2U);
  float*    r2v     = (float*)(ws + O_R2V);
  float*    offs    = (float*)(ws + O_OFFS);
  float*    attn    = (float*)(ws + O_ATTN);
  float*    outatt  = (float*)(ws + O_OUTATT);
  _Float16* outh16  = (_Float16*)(ws + O_OUTH16);
  float*    attnout = (float*)(ws + O_ATTOUT);
  float*    q2      = (float*)(ws + O_Q2);
  _Float16* q2h     = (_Float16*)(ws + O_Q2H);
  float*    hbuf    = (float*)(ws + O_HBUF);
  _Float16* hbuf16  = (_Float16*)(ws + O_HBUF16);
  float*    ffn     = (float*)(ws + O_FFN);
  _Float16* q3h     = (_Float16*)(ws + O_Q3H);
  float*    tbuf    = (float*)(ws + O_TBUF);
  _Float16* tbuf16  = (_Float16*)(ws + O_TBUF16);
  float*    out_q   = (float*)d_out;                 // 1024*128 query
  float*    out_p   = (float*)d_out + NQ * CCH;      // 1024*28 props

  auto GEMM = [&](const _Float16* A, const _Float16* Bt, const float* bias, float* C,
                  _Float16* C16, int M, int N, int K, int act) {
    dim3 g(M / 16, N / 64);
    k_wmma_gemm<<<g, 128, 0, stream>>>(A, Bt, bias, C, C16, M, N, K, act);
  };
  auto TRANS = [&](const float* s, _Float16* d, int K, int N) {
    int tot = K * N;
    k_transpose_f16<<<(tot + 255) / 256, 256, 0, stream>>>(s, d, K, N);
  };

  // --- 1. voxel scatter-mean ---
  long long zn = (long long)GRID3 * CCH + GRID3;
  k_fill_zero<<<(int)((zn + 255) / 256), 256, 0, stream>>>(grid_f, zn);
  k_scatter<<<NQ, CCH, 0, stream>>>(q, gprops, grid_f, cnt_f);

  // --- 2. sparse 3x3x3 conv as gathered GEMM ---
  k_gather_convA<<<NQ, CCH, 0, stream>>>(gprops, grid_f, cnt_f, Aconv);
  TRANS(Wsc, wsct, KCONV, CCH);
  TRANS(offw, offwt, CCH, 2048);
  TRANS(aww, awwt, CCH, 1024);
  TRANS(valw, valwt, CCH, CCH);
  TRANS(outw, outwt, CCH, CCH);
  TRANS(f1w, f1t, CCH, 512);
  TRANS(f2w, f2t, 512, CCH);
  TRANS(p1w, p1t, CCH, CCH);
  GEMM(Aconv, wsct, bsc, convout, nullptr, NQ, CCH, KCONV, /*gelu*/1);
  k_layernorm_res<<<NQ, CCH, 0, stream>>>(q, convout, scg, scb, q1, q1h);

  // --- 3. value projection (HBM-streaming GEMM) ---
  long long msn = (long long)MSROWS * CCH;
  k_f32_to_f16<<<(int)((msn + 255) / 256), 256, 0, stream>>>(msf, msf16, msn);
  GEMM(msf16, valwt, valb, value, nullptr, MSROWS, CCH, CCH, 0);

  // --- 4. per-query projections ---
  GEMM(q1h, offwt, offb, offs, nullptr, NQ, 2048, CCH, 0);
  GEMM(q1h, awwt, awb, attn, nullptr, NQ, 1024, CCH, 0);
  k_softmax128<<<NQ * 8, CCH, 0, stream>>>(attn);
  k_refoff<<<NQ, 32, 0, stream>>>(q1, row, rob, refoff);
  k_geometry<<<(NCAM * NQ * 4 + 255) / 256, 256, 0, stream>>>(gprops, refoff, l2i, depth,
                                                              wmask, r2u, r2v);

  // --- 5. deformable attention sampling ---
  k_deform_sample<<<NQ, CCH, 0, stream>>>(value, attn, offs, wmask, r2u, r2v, outatt);
  k_f32_to_f16<<<(NQ * CCH + 255) / 256, 256, 0, stream>>>(outatt, outh16, NQ * CCH);
  GEMM(outh16, outwt, outb, attnout, nullptr, NQ, CCH, CCH, 0);
  k_layernorm_res<<<NQ, CCH, 0, stream>>>(q1, attnout, ang, anb, q2, q2h);

  // --- 6. FFN ---
  GEMM(q2h, f1t, f1b, hbuf, hbuf16, NQ, 512, CCH, /*gelu*/1);
  GEMM(hbuf16, f2t, f2b, ffn, nullptr, NQ, CCH, 512, 0);
  k_layernorm_res<<<NQ, CCH, 0, stream>>>(q2, ffn, fng, fnb, out_q, q3h);

  // --- 7. props head ---
  GEMM(q3h, p1t, p1b, tbuf, tbuf16, NQ, CCH, CCH, /*gelu*/1);
  k_props_head<<<NQ, 32, 0, stream>>>(tbuf, p2w, p2b, gprops, out_p);
}
